// GNNEncoder_84670985273757
// MI455X (gfx1250) — compile-verified, matrix operand order checked by reading, not run on hardware
//
#include <hip/hip_runtime.h>
#include <hip/hip_bf16.h>

// ---------------------------------------------------------------------------
// Hetero-GraphSAGE (2 layers, 3 node types, 4 relations) for MI455X / gfx1250.
// Scatter-mean in f32 with native global_atomic_add_f32; GEMMs via
// v_wmma_f32_16x16x32_bf16 (wave32), f32 accumulation, dual independent
// accumulator chains to avoid WMMA->WMMA RAW hazard NOPs.
// ---------------------------------------------------------------------------

typedef __bf16 bf16_t;
typedef __attribute__((ext_vector_type(8)))  __bf16 v8bf;
typedef __attribute__((ext_vector_type(16))) __bf16 v16bf;
typedef __attribute__((ext_vector_type(8)))  float  v8f;

#define NN 100000     // nodes per type
#define EE 600000     // edges per relation
#define FD 128        // feature dim (F == H == O == 128)
#define NF (NN * FD)  // floats per node-feature matrix
#define MTILES (NN / 16)  // 6250 row tiles, exact

// ---------------------------------------------------------------------- utils
__device__ __forceinline__ void atomic_add_f32(float* p, float v) {
  // Lowers to global_atomic_add_f32 (no CAS loop) — coarse-grained memory.
  unsafeAtomicAdd(p, v);
}

// -------------------------------------------------------------- f32 -> bf16
__global__ void k_f32_to_bf16(const float* __restrict__ x,
                              bf16_t* __restrict__ y, int n) {
  int i = blockIdx.x * blockDim.x + threadIdx.x;
  if (i < n) y[i] = (bf16_t)x[i];
}

// ----------------------------------------------------- edge scatter (gather+add)
// One block per edge, 128 threads = one feature row. Coalesced gather,
// native f32 atomic scatter-add. T is float (layer 0) or bf16 (layer 1).
template <typename T>
__global__ void k_scatter(const T* __restrict__ xsrc,
                          const int* __restrict__ src,
                          const int* __restrict__ dst,
                          float* __restrict__ msg,
                          float* __restrict__ cnt) {
  int e = blockIdx.x;
  int f = threadIdx.x;
  int s = src[e];
  int d = dst[e];
  float v = (float)xsrc[(size_t)s * FD + f];
  atomic_add_f32(&msg[(size_t)d * FD + f], v);
  if (f == 0) atomic_add_f32(&cnt[d], 1.0f);
}

// ------------------------------------------------- mean = msg / max(cnt,1) -> bf16
__global__ void k_mean_bf16(const float* __restrict__ msg,
                            const float* __restrict__ cnt,
                            bf16_t* __restrict__ out, int n) {
  int i = blockIdx.x * blockDim.x + threadIdx.x;
  if (i >= n) return;
  float c = cnt[i >> 7];
  out[i] = (bf16_t)(msg[i] / fmaxf(c, 1.0f));
}

// ------------------------------------------- pack 128x128 f32 W into WMMA B layout
// B is 32x16 (KxN) per wmma. 16-bit B layout (wave32):
//   lane 0..15  -> col N=lane,     K = kbase..kbase+7 (elems 0..7),
//                                  K = kbase+16..+23 (elems 8..15), kbase=0
//   lane 16..31 -> col N=lane-16,  same but kbase=8
// Packed so each lane's 16 bf16 are contiguous (one 32B load in the GEMM).
__global__ void k_pack_w(const float* __restrict__ W, bf16_t* __restrict__ pk) {
  int nt = blockIdx.x;      // 0..7   (N tile)
  int kc = blockIdx.y;      // 0..3   (K chunk of 32)
  int lane = threadIdx.x;   // 0..31
  int col = nt * 16 + (lane & 15);
  int kbase = (lane < 16) ? 0 : 8;
  bf16_t* dst = pk + ((((nt << 2) + kc) * 32 + lane) << 4);
#pragma unroll
  for (int e = 0; e < 8; ++e)
    dst[e] = (bf16_t)W[(kc * 32 + kbase + e) * FD + col];
#pragma unroll
  for (int e = 0; e < 8; ++e)
    dst[8 + e] = (bf16_t)W[(kc * 32 + 16 + kbase + e) * FD + col];
}

// --------------------------------------------------------------- A tile loader
// 16-bit A layout (16x32, row M = lane&15): elems 0..7 = K kbase..kbase+7,
// elems 8..15 = K kbase+16..kbase+23 (kbase = 0 / 8 by lane half). Both halves
// are contiguous 16B runs in the row-major bf16 feature matrix.
__device__ __forceinline__ v16bf load_a(const bf16_t* p) {
  v8bf lo = *reinterpret_cast<const v8bf*>(p);
  v8bf hi = *reinterpret_cast<const v8bf*>(p + 16);
  return __builtin_shufflevector(lo, hi, 0, 1, 2, 3, 4, 5, 6, 7, 8, 9, 10, 11,
                                 12, 13, 14, 15);
}

// ------------------------------------------------------------------ fused GEMM
// Y[16x128 tile] += A1 @ W1 + A2 @ W2 + bias   (A* are Nx128 bf16, W* packed)
// grid.x = MTILES, block = 256 (8 waves); wave w owns output cols [16w,16w+16).
// Two independent accumulator chains (c1: A1@W1 + Y + bias, c2: A2@W2) so the
// scheduler can interleave WMMAs without the D->C back-to-back hazard NOPs.
__global__ __launch_bounds__(256) void k_gemm_acc(
    float* __restrict__ Y, const bf16_t* __restrict__ A1,
    const bf16_t* __restrict__ A2, const bf16_t* __restrict__ W1pk,
    const bf16_t* __restrict__ W2pk, const float* __restrict__ bias) {
  int mt = blockIdx.x;
  int wave = threadIdx.x >> 5;
  int lane = threadIdx.x & 31;
  int col = wave * 16 + (lane & 15);
  int rowA = mt * 16 + (lane & 15);
  int kbase = (lane < 16) ? 0 : 8;
  int rowC0 = mt * 16 + ((lane < 16) ? 0 : 8);  // C: VGPR i -> row rowC0+i

  float bv = bias[col];
  v8f c1;  // chain 1: Y + bias + A1@W1
  v8f c2;  // chain 2: A2@W2
#pragma unroll
  for (int i = 0; i < 8; ++i) {
    c1[i] = Y[(size_t)(rowC0 + i) * FD + col] + bv;
    c2[i] = 0.0f;
  }

#pragma unroll
  for (int kc = 0; kc < 4; ++kc) {
    v16bf a1 = load_a(A1 + (size_t)rowA * FD + kc * 32 + kbase);
    v16bf b1 = *reinterpret_cast<const v16bf*>(
        W1pk + ((((wave << 2) + kc) * 32 + lane) << 4));
    c1 = __builtin_amdgcn_wmma_f32_16x16x32_bf16(false, a1, false, b1, (short)0,
                                                 c1, false, false);
    v16bf a2 = load_a(A2 + (size_t)rowA * FD + kc * 32 + kbase);
    v16bf b2 = *reinterpret_cast<const v16bf*>(
        W2pk + ((((wave << 2) + kc) * 32 + lane) << 4));
    c2 = __builtin_amdgcn_wmma_f32_16x16x32_bf16(false, a2, false, b2, (short)0,
                                                 c2, false, false);
  }

#pragma unroll
  for (int i = 0; i < 8; ++i)
    Y[(size_t)(rowC0 + i) * FD + col] = c1[i] + c2[i];
}

// -------------------------------------------------------------------- ReLUs
__global__ void k_lrelu_to_bf16(const float* __restrict__ y,
                                bf16_t* __restrict__ xb, int n) {
  int i = blockIdx.x * blockDim.x + threadIdx.x;
  if (i >= n) return;
  float v = y[i];
  xb[i] = (bf16_t)(v > 0.0f ? v : 0.01f * v);
}

__global__ void k_lrelu_inplace(float* __restrict__ y, int n) {
  int i = blockIdx.x * blockDim.x + threadIdx.x;
  if (i >= n) return;
  float v = y[i];
  y[i] = (v > 0.0f ? v : 0.01f * v);
}

// ---------------------------------------------------------------------------
extern "C" void kernel_launch(void* const* d_in, const int* in_sizes, int n_in,
                              void* d_out, int out_size, void* d_ws,
                              size_t ws_size, hipStream_t stream) {
  (void)in_sizes; (void)n_in; (void)out_size; (void)ws_size;

  const float* x_in[3] = {(const float*)d_in[0], (const float*)d_in[1],
                          (const float*)d_in[2]};
  const int* ei[4] = {(const int*)d_in[3], (const int*)d_in[4],
                      (const int*)d_in[5], (const int*)d_in[6]};
  // relation r: src node type, dst node type (0-based: x1,x2,x3)
  const int rsrc[4] = {1, 0, 2, 1};  // p23:x2  p12:x1  p32:x3  p21:x2
  const int rdst[4] = {2, 1, 1, 0};  // ->x3    ->x2    ->x2    ->x1

  // ---- workspace carve (256B aligned) --------------------------------
  char* base = (char*)d_ws;
  size_t off = 0;
  auto carve = [&](size_t bytes) -> void* {
    void* p = base + off;
    off = (off + bytes + 255) & ~(size_t)255;
    return p;
  };
  float*  msg   = (float*)carve((size_t)NF * 4);
  float*  cnt   = (float*)carve((size_t)NN * 4);
  bf16_t* meanb = (bf16_t*)carve((size_t)NF * 2);
  bf16_t* xb[3];
  for (int t = 0; t < 3; ++t) xb[t] = (bf16_t*)carve((size_t)NF * 2);
  bf16_t* wlpk = (bf16_t*)carve((size_t)FD * FD * 2);
  bf16_t* wrpk = (bf16_t*)carve((size_t)FD * FD * 2);

  // Y accumulators live in d_out: [Y1 | Y2 | Y3]
  float* Y[3] = {(float*)d_out, (float*)d_out + (size_t)NF,
                 (float*)d_out + 2 * (size_t)NF};

  const int TB = 256;
  const int gNF = (NF + TB - 1) / TB;

  // layer-0 inputs as bf16 (for x_dst @ Wr term)
  for (int t = 0; t < 3; ++t)
    k_f32_to_bf16<<<gNF, TB, 0, stream>>>(x_in[t], xb[t], NF);

  for (int l = 0; l < 2; ++l) {
    hipMemsetAsync(d_out, 0, 3 * (size_t)NF * sizeof(float), stream);

    for (int r = 0; r < 4; ++r) {
      // -------- segment mean (gather + scatter-add + divide) --------
      hipMemsetAsync(msg, 0, (size_t)NF * sizeof(float), stream);
      hipMemsetAsync(cnt, 0, (size_t)NN * sizeof(float), stream);
      const int* src = ei[r];
      const int* dst = ei[r] + EE;
      if (l == 0)
        k_scatter<float><<<EE, FD, 0, stream>>>(x_in[rsrc[r]], src, dst, msg,
                                                cnt);
      else
        k_scatter<bf16_t><<<EE, FD, 0, stream>>>(xb[rsrc[r]], src, dst, msg,
                                                 cnt);
      k_mean_bf16<<<gNF, TB, 0, stream>>>(msg, cnt, meanb, NF);

      // -------- weights -> WMMA-packed bf16 --------
      int pb = 7 + (l * 4 + r) * 3;  // (Wl, bl, Wr)
      k_pack_w<<<dim3(8, 4), 32, 0, stream>>>((const float*)d_in[pb], wlpk);
      k_pack_w<<<dim3(8, 4), 32, 0, stream>>>((const float*)d_in[pb + 2], wrpk);

      // -------- Y_dst += mean @ Wl + x_dst @ Wr + bias --------
      k_gemm_acc<<<MTILES, 256, 0, stream>>>(Y[rdst[r]], meanb, xb[rdst[r]],
                                             wlpk, wrpk,
                                             (const float*)d_in[pb + 1]);
    }

    if (l == 0) {
      // activations for next layer (bf16 feeds both aggregation and GEMM)
      for (int t = 0; t < 3; ++t)
        k_lrelu_to_bf16<<<gNF, TB, 0, stream>>>(Y[t], xb[t], NF);
    } else {
      k_lrelu_inplace<<<(3 * NF + TB - 1) / TB, TB, 0, stream>>>((float*)d_out,
                                                                 3 * NF);
    }
  }
}
